// GraphConvolution_57947698758288
// MI455X (gfx1250) — compile-verified, matrix operand order checked by reading, not run on hardware
//
#include <hip/hip_runtime.h>
#include <hip/hip_bf16.h>

// GraphConvolution: out = elu(segment_sum(vals * (xW+b)[cols], rows))
// MI455X plan:
//   K1: memset d_out = 0
//   K2: h = x@W + b via v_wmma_f32_16x16x32_f16 (f16 in, f32 acc), h -> d_ws
//   K3: edge gather (float4, L2-resident h) + global_atomic_add_f32 scatter
//   K4: ELU in place on d_out

typedef __attribute__((ext_vector_type(16))) _Float16 v16h;
typedef __attribute__((ext_vector_type(8)))  float    v8f;

#define GC_NIN  256
#define GC_NOUT 64
#define LDS_KPAD (GC_NIN + 8)   // pad LDS rows to dodge bank conflicts

// ---------------------------------------------------------------------------
// GEMM: h = x @ W + b.  One wave32 per 16-row band of x; 4 WMMA column tiles.
// ---------------------------------------------------------------------------
__global__ void __launch_bounds__(256)
gc_gemm_wmma(const float* __restrict__ x,
             const float* __restrict__ W,     // [256][64] row-major
             const float* __restrict__ bias,  // [64]
             float* __restrict__ h,           // [n_nodes][64]
             int n_nodes)
{
    // W transposed into LDS as f16: Wt[n][k]
    __shared__ _Float16 Wt[GC_NOUT][LDS_KPAD];
    for (int idx = threadIdx.x; idx < GC_NIN * GC_NOUT; idx += blockDim.x) {
        int k = idx >> 6;          // / 64
        int n = idx & 63;
        Wt[n][k] = (_Float16)W[idx];
    }
    __syncthreads();

    const int wave = threadIdx.x >> 5;   // wave32
    const int lane = threadIdx.x & 31;
    const int m0   = (blockIdx.x * 8 + wave) * 16;
    if (m0 >= n_nodes) return;

    // Per-lane fragment coordinates (CDNA5 16-bit A / f32 C layouts):
    const int mrow  = m0 + (lane & 15);          // A: row this lane feeds
    const int kbase = (lane < 16) ? 0 : 8;       // A/B: K sub-offset per half-wave
    const int ncol  = lane & 15;                 // B/C/D: column within 16-wide tile

    v8f acc0 = {}, acc1 = {}, acc2 = {}, acc3 = {};

    const float* xrow = x + (size_t)mrow * GC_NIN;

    for (int kk = 0; kk < GC_NIN; kk += 32) {
        const float* xr = xrow + kk + kbase;
        if (kk + 32 < GC_NIN)
            __builtin_prefetch(xr + 32, 0, 1);   // global_prefetch next K tile

        // A fragment: VGPRs 0..3 hold K = kbase+0..7, VGPRs 4..7 hold K = kbase+16..23
        v16h a;
        #pragma unroll
        for (int i = 0; i < 8; ++i) a[i]     = (_Float16)xr[i];
        #pragma unroll
        for (int i = 0; i < 8; ++i) a[8 + i] = (_Float16)xr[16 + i];

        // Load ALL four B fragments first (8x ds_load_b128 as one clause,
        // single dscnt wait), then fire the 4 WMMAs back-to-back.
        v16h bf0, bf1, bf2, bf3;
        {
            const _Float16* wp0 = &Wt[ 0 + ncol][kk + kbase];
            const _Float16* wp1 = &Wt[16 + ncol][kk + kbase];
            const _Float16* wp2 = &Wt[32 + ncol][kk + kbase];
            const _Float16* wp3 = &Wt[48 + ncol][kk + kbase];
            #pragma unroll
            for (int i = 0; i < 8; ++i) {
                bf0[i] = wp0[i];  bf0[8 + i] = wp0[16 + i];
                bf1[i] = wp1[i];  bf1[8 + i] = wp1[16 + i];
                bf2[i] = wp2[i];  bf2[8 + i] = wp2[16 + i];
                bf3[i] = wp3[i];  bf3[8 + i] = wp3[16 + i];
            }
        }

        acc0 = __builtin_amdgcn_wmma_f32_16x16x32_f16(false, a, false, bf0,
                                                      (short)0, acc0, false, false);
        acc1 = __builtin_amdgcn_wmma_f32_16x16x32_f16(false, a, false, bf1,
                                                      (short)0, acc1, false, false);
        acc2 = __builtin_amdgcn_wmma_f32_16x16x32_f16(false, a, false, bf2,
                                                      (short)0, acc2, false, false);
        acc3 = __builtin_amdgcn_wmma_f32_16x16x32_f16(false, a, false, bf3,
                                                      (short)0, acc3, false, false);
    }

    // D layout: VGPR r -> M = m0 + r + 8*(lane>=16), N = t*16 + ncol
    const int rbase = (lane >> 4) * 8;
    #pragma unroll
    for (int t = 0; t < 4; ++t) {
        const v8f c = (t == 0) ? acc0 : (t == 1) ? acc1 : (t == 2) ? acc2 : acc3;
        const int n = t * 16 + ncol;
        const float bn = bias[n];
        #pragma unroll
        for (int r = 0; r < 8; ++r) {
            int m = m0 + rbase + r;
            h[(size_t)m * GC_NOUT + n] = c[r] + bn;
        }
    }
}

// ---------------------------------------------------------------------------
// Edge scatter: 16 threads per edge; float4 gather of h[col], f32 atomics to out[row]
// ---------------------------------------------------------------------------
__global__ void __launch_bounds__(256)
gc_edge_scatter(const int* __restrict__ rows,
                const int* __restrict__ cols,
                const float* __restrict__ vals,
                const float* __restrict__ h,
                float* __restrict__ out,
                int n_edges)
{
    long long t = (long long)blockIdx.x * blockDim.x + threadIdx.x;
    int e = (int)(t >> 4);
    if (e >= n_edges) return;
    int l16 = (int)(t & 15);

    float v = vals[e];
    int c = cols[e];
    int r = rows[e];

    // 16 lanes * float4 == one 256B h row, fully coalesced; h is L2-resident
    float4 s = ((const float4*)(h + (size_t)c * GC_NOUT))[l16];

    float* dst = out + (size_t)r * GC_NOUT + l16 * 4;
    atomicAdd(dst + 0, v * s.x);
    atomicAdd(dst + 1, v * s.y);
    atomicAdd(dst + 2, v * s.z);
    atomicAdd(dst + 3, v * s.w);
}

// ---------------------------------------------------------------------------
// ELU in place (alpha = 1)
// ---------------------------------------------------------------------------
__global__ void __launch_bounds__(256)
gc_elu_inplace(float* __restrict__ out, int n4)
{
    int i = blockIdx.x * blockDim.x + threadIdx.x;
    if (i >= n4) return;
    float4 v = ((const float4*)out)[i];
    v.x = (v.x > 0.f) ? v.x : (expf(v.x) - 1.f);
    v.y = (v.y > 0.f) ? v.y : (expf(v.y) - 1.f);
    v.z = (v.z > 0.f) ? v.z : (expf(v.z) - 1.f);
    v.w = (v.w > 0.f) ? v.w : (expf(v.w) - 1.f);
    ((float4*)out)[i] = v;
}

// ---------------------------------------------------------------------------
extern "C" void kernel_launch(void* const* d_in, const int* in_sizes, int n_in,
                              void* d_out, int out_size, void* d_ws, size_t ws_size,
                              hipStream_t stream)
{
    const float* x    = (const float*)d_in[0];
    const int*   erow = (const int*)  d_in[1];
    const int*   ecol = (const int*)  d_in[2];
    const float* eval = (const float*)d_in[3];
    const float* W    = (const float*)d_in[4];
    const float* bias = (const float*)d_in[5];

    float* out = (float*)d_out;
    float* h   = (float*)d_ws;          // [n_nodes][64] fp32 scratch (25.6 MB)

    const int n_nodes = in_sizes[0] / GC_NIN;   // 100000
    const int n_edges = in_sizes[1];            // 1600000

    // Zero accumulators (d_out is poisoned by the harness)
    hipMemsetAsync(d_out, 0, (size_t)n_nodes * GC_NOUT * sizeof(float), stream);

    // GEMM: one wave per 16-row tile, 8 waves per block
    const int row_tiles   = (n_nodes + 15) / 16;
    const int gemm_blocks = (row_tiles + 7) / 8;
    gc_gemm_wmma<<<gemm_blocks, 256, 0, stream>>>(x, W, bias, h, n_nodes);

    // Edge gather + atomic scatter: 16 threads per edge
    const long long eth = (long long)n_edges * 16;
    const int eblocks = (int)((eth + 255) / 256);
    gc_edge_scatter<<<eblocks, 256, 0, stream>>>(erow, ecol, eval, h, out, n_edges);

    // ELU epilogue
    const int n4 = n_nodes * GC_NOUT / 4;
    gc_elu_inplace<<<(n4 + 255) / 256, 256, 0, stream>>>(out, n4);
}
